// RNNWithVectorSelection_54425825575654
// MI455X (gfx1250) — compile-verified, matrix operand order checked by reading, not run on hardware
//
#include <hip/hip_runtime.h>
#include <hip/hip_bf16.h>

typedef __attribute__((ext_vector_type(16))) __bf16 v16bf;
typedef __attribute__((ext_vector_type(8)))  __bf16 v8bf;
typedef __attribute__((ext_vector_type(8)))  float  v8f;
typedef __attribute__((ext_vector_type(4)))  unsigned int uint4v;

constexpr int kB = 32, kT = 256, kE = 512, kS = 5, kV = 32000;
constexpr int kRows = kB * kT;             // 8192
constexpr int kNChunk = 256;               // vocab cols per workgroup (LDS-resident)
constexpr int kNumChunks = kV / kNChunk;   // 125
constexpr int kNPart = kV / 64;            // 500 partial sets (64-col per wave)

// ---------------- wave32 reductions ----------------
__device__ __forceinline__ float wave_sum(float v) {
  #pragma unroll
  for (int m = 16; m >= 1; m >>= 1) v += __shfl_xor(v, m, 32);
  return v;
}
__device__ __forceinline__ float half_sum16(float v) {
  #pragma unroll
  for (int m = 8; m >= 1; m >>= 1) v += __shfl_xor(v, m, 32);
  return v;
}
__device__ __forceinline__ float half_max16(float v) {
  #pragma unroll
  for (int m = 8; m >= 1; m >>= 1) v = fmaxf(v, __shfl_xor(v, m, 32));
  return v;
}

// ---------------- WMMA helpers ----------------
union Frag16 { v16bf v; v8bf h[2]; };

// A/B fragment for 16x16x32 bf16 WMMA. ISA layout (16-bit A 16x32):
// lane L: row = L%16 ; lanes<16 cover K = [k0, k0+8) and [k0+16, k0+24),
// lanes>=16 cover K = [k0+8, k0+16) and [k0+24, k0+32).
// Caller passes p = base + k0 + ((lane>>4)<<3); we load 8 halves at p and p+16.
__device__ __forceinline__ v16bf load_frag(const __bf16* p) {
  Frag16 f;
  f.h[0] = *(const v8bf*)(p);
  f.h[1] = *(const v8bf*)(p + 16);
  return f.v;
}

__device__ __forceinline__ v8f wmma_bf16(v16bf a, v16bf b, v8f c) {
  return __builtin_amdgcn_wmma_f32_16x16x32_bf16(false, a, false, b, (short)0, c,
                                                 false, false);
}

// ---------------- K1: gather + normalize latent rows ----------------
// lat = latent[zi].reshape(B,S,E); lat /= (1e-5 + std(lat, ddof=1)); lat *= 0.113
__global__ void k_gather_norm(const int* __restrict__ zi,
                              const float* __restrict__ latent,
                              float* __restrict__ latn) {
  const int row = blockIdx.x;          // 0..B*S-1
  const int b = row / kS, s = row % kS;
  const int src = zi[b];
  const float* x = latent + (size_t)src * (kS * kE) + (size_t)s * kE;
  float* o = latn + (size_t)b * (kS * kE) + (size_t)s * kE;

  float sx = 0.f, sx2 = 0.f;
  for (int e = threadIdx.x; e < kE; e += blockDim.x) {
    float v = x[e]; sx += v; sx2 += v * v;
  }
  __shared__ float rs[16], rs2[16];
  sx = wave_sum(sx); sx2 = wave_sum(sx2);
  const int w = threadIdx.x >> 5, lane = threadIdx.x & 31;
  if (lane == 0) { rs[w] = sx; rs2[w] = sx2; }
  __syncthreads();
  if (threadIdx.x == 0) {
    float a = 0.f, a2 = 0.f;
    const int nw = blockDim.x >> 5;
    for (int i = 0; i < nw; ++i) { a += rs[i]; a2 += rs2[i]; }
    float mean = a * (1.0f / kE);
    float var = (a2 - (float)kE * mean * mean) * (1.0f / (kE - 1));
    rs[0] = 0.113f / (1e-5f + sqrtf(fmaxf(var, 0.f)));
  }
  __syncthreads();
  const float c = rs[0];
  for (int e = threadIdx.x; e < kE; e += blockDim.x) o[e] = c * x[e];
}

// ---------------- K2a: split f32 -> bf16 hi/lo (for bf16x3 GEMM) ----------------
__global__ void k_split_bf16(const float* __restrict__ src, __bf16* __restrict__ hi,
                             __bf16* __restrict__ lo, int n) {
  int i = blockIdx.x * blockDim.x + threadIdx.x;
  if (i < n) {
    float v = src[i];
    __bf16 h = (__bf16)v;
    hi[i] = h;
    lo[i] = (__bf16)(v - (float)h);
  }
}

// ---------------- K2b: f32 -> bf16 ----------------
__global__ void k_to_bf16(const float* __restrict__ src, __bf16* __restrict__ dst,
                          size_t n) {
  size_t i = blockIdx.x * (size_t)blockDim.x + threadIdx.x;
  for (; i < n; i += (size_t)gridDim.x * blockDim.x) dst[i] = (__bf16)src[i];
}

// ---------------- K3: sequential recurrence (2 blocks x 16 batches) ----------------
// per step: zn = normalize(z); att = zn@W_ext.T + b_ext; emit = att@[zn, lat1..4];
//           z  = (zn + zn@W_tr.T + b_tr)/2   (bf16x3 WMMA, M=16 N=512 K=512)
constexpr int kRecLds = 242208;
__global__ void __launch_bounds__(512)
k_recurrence(const float* __restrict__ latn,
             const __bf16* __restrict__ wtr_hi,
             const __bf16* __restrict__ wtr_lo,
             const float* __restrict__ b_tr,
             const float* __restrict__ w_ext,
             const float* __restrict__ b_ext,
             float* __restrict__ zs_f32,
             __bf16* __restrict__ zs_bf) {
  extern __shared__ char smem[];
  float*  sZ   = (float*)(smem);               // 16 x 512 current z
  float*  sZn  = (float*)(smem + 32768);       // 16 x 512 normalized zn (f32)
  __bf16* sH   = (__bf16*)(smem + 65536);      // 16 x 520 zn hi (padded rows)
  __bf16* sL   = (__bf16*)(smem + 82176);      // 16 x 520 zn lo
  float*  sLat = (float*)(smem + 98816);       // 16 x 4 x 512 lat rows s=1..4
  float*  sWx  = (float*)(smem + 229888);      // 5 x 512
  float*  sBtr = (float*)(smem + 240128);      // 512
  float*  sBex = (float*)(smem + 242176);      // 5

  const int tid = threadIdx.x;
  const int w = tid >> 5, lane = tid & 31;     // wave w owns batch w (16 waves)
  const int bbase = blockIdx.x * 16;

  for (int i = tid; i < 16 * kE; i += blockDim.x) {
    int bb = i >> 9, e = i & 511;
    sZ[i] = latn[(size_t)(bbase + bb) * (kS * kE) + e];       // z_init = lat[:, 0]
  }
  for (int i = tid; i < 16 * 4 * kE; i += blockDim.x) {
    int bb = i >> 11, rest = i & 2047, s = rest >> 9, e = rest & 511;
    sLat[i] = latn[(size_t)(bbase + bb) * (kS * kE) + (size_t)(s + 1) * kE + e];
  }
  for (int i = tid; i < 5 * kE; i += blockDim.x) sWx[i] = w_ext[i];
  for (int i = tid; i < kE; i += blockDim.x) sBtr[i] = b_tr[i];
  if (tid < 5) sBex[tid] = b_ext[tid];
  __syncthreads();

  for (int t = 0; t < kT; ++t) {
    // ---- stage 1 (wave-private per batch): stats, normalize, att, emit ----
    {
      const float* zr = sZ + w * kE;
      float sx = 0.f, sx2 = 0.f;
      for (int e = lane; e < kE; e += 32) { float x = zr[e]; sx += x; sx2 += x * x; }
      sx = wave_sum(sx); sx2 = wave_sum(sx2);
      float mean = sx * (1.0f / kE);
      float var = (sx2 - (float)kE * mean * mean) * (1.0f / (kE - 1));
      float c = 0.113f / (1e-5f + sqrtf(fmaxf(var, 0.f)));

      float* znr = sZn + w * kE;
      __bf16* hr = sH + w * 520;
      __bf16* lr = sL + w * 520;
      for (int e = lane; e < kE; e += 32) {
        float v = c * zr[e];
        znr[e] = v;
        __bf16 h = (__bf16)v;
        hr[e] = h;
        lr[e] = (__bf16)(v - (float)h);
      }
      float att[5];
      #pragma unroll
      for (int s = 0; s < 5; ++s) {
        const float* wr = sWx + s * kE;
        float d = 0.f;
        for (int e = lane; e < kE; e += 32) d += znr[e] * wr[e];
        att[s] = wave_sum(d) + sBex[s];
      }
      const float* l1 = sLat + w * 2048;
      float*  zo = zs_f32 + ((size_t)(bbase + w) * kT + t) * kE;
      __bf16* zb = zs_bf + ((size_t)(bbase + w) * kT + t) * kE;
      for (int e = lane; e < kE; e += 32) {
        float v = att[0] * znr[e] + att[1] * l1[e] + att[2] * l1[512 + e] +
                  att[3] * l1[1024 + e] + att[4] * l1[1536 + e];
        zo[e] = v;
        zb[e] = (__bf16)v;
      }
    }
    __syncthreads();
    // ---- stage 2: z = (zn + zn @ Wtr^T + b)/2, bf16x3 WMMA; wave w does 2 N-tiles ----
    {
      const int mrow = lane & 15;
      const int hof = (lane >> 4) << 3;    // 0 or 8
      #pragma unroll 1
      for (int tile = 0; tile < 2; ++tile) {
        const int n0 = (w * 2 + tile) * 16;
        const int n = n0 + mrow;           // lane's B column
        v8f acc = {};
        const __bf16* bhi = wtr_hi + (size_t)n * kE;   // B[k][n] = W_tr[n][k]
        const __bf16* blo = wtr_lo + (size_t)n * kE;
        const __bf16* aph = sH + mrow * 520 + hof;
        const __bf16* apl = sL + mrow * 520 + hof;
        // manual x2 unroll, no compiler unroll: bounded liveness, pipelined loads
        #pragma unroll 1
        for (int k0 = 0; k0 < kE; k0 += 64) {
          v16bf ah0 = load_frag(aph + k0);
          v16bf al0 = load_frag(apl + k0);
          v16bf bh0 = load_frag(bhi + k0 + hof);
          v16bf bl0 = load_frag(blo + k0 + hof);
          v16bf ah1 = load_frag(aph + k0 + 32);
          v16bf al1 = load_frag(apl + k0 + 32);
          v16bf bh1 = load_frag(bhi + k0 + 32 + hof);
          v16bf bl1 = load_frag(blo + k0 + 32 + hof);
          acc = wmma_bf16(ah0, bh0, acc);
          acc = wmma_bf16(ah0, bl0, acc);
          acc = wmma_bf16(al0, bh0, acc);
          acc = wmma_bf16(ah1, bh1, acc);
          acc = wmma_bf16(ah1, bl1, acc);
          acc = wmma_bf16(al1, bh1, acc);
        }
        // D layout: vgpr r -> M = r (lanes 0-15) or r+8 (lanes 16-31); N = n
        #pragma unroll
        for (int r = 0; r < 8; ++r) {
          int mm = r + hof;
          sZ[mm * kE + n] = 0.5f * (sZn[mm * kE + n] + acc[r] + sBtr[n]);
        }
      }
    }
    __syncthreads();
  }
}

// ---------------- K4: vocab GEMM + per-64-col logsumexp partials ----------------
// grid = 2 m-halves x 125 chunks; block = 256 thr (8 waves as 2m x 4n, 64x64/wave)
constexpr int kVocLds = 266240 + 128 * 144;    // B chunk + A k-slice
__global__ void __launch_bounds__(256)
k_vocab_lse(const __bf16* __restrict__ zsb,
            const __bf16* __restrict__ wvb,
            const float* __restrict__ b_voc,
            float* __restrict__ pm, float* __restrict__ ps) {
  extern __shared__ char smem[];
  __bf16* ldsB = (__bf16*)smem;              // 256 x (512+8) halves
  __bf16* ldsA = (__bf16*)(smem + 266240);   // 128 x (64+8) halves
  const int tid = threadIdx.x;
  const int w = tid >> 5, lane = tid & 31;
  const int chunk = blockIdx.x % kNumChunks;
  const int mhalf = blockIdx.x / kNumChunks;
  const int nbase = chunk * kNChunk;
  const int wm = w & 1;          // m position (0..1)
  const int wn = w >> 1;         // n position (0..3)

  // resident B chunk: W_vocab[nbase+n][k], row pitch 520 halves (bank stagger)
  for (int c = tid; c < kNChunk * 64; c += blockDim.x) {
    int n = c >> 6, kc = c & 63;
    *(uint4v*)((char*)ldsB + n * 1040 + kc * 16) =
        *(const uint4v*)((const char*)(wvb + (size_t)(nbase + n) * kE) + kc * 16);
  }
  float bv[4];
  #pragma unroll
  for (int ni = 0; ni < 4; ++ni)
    bv[ni] = b_voc[nbase + wn * 64 + ni * 16 + (lane & 15)];
  __syncthreads();

  const int mrow = lane & 15;
  const int hof = (lane >> 4) << 3;

  #pragma unroll 1
  for (int mb = mhalf * 32; mb < mhalf * 32 + 32; ++mb) {
    const int rowbase = mb * 128;
    v8f acc[4][4];
    v8f zero = {0.f, 0.f, 0.f, 0.f, 0.f, 0.f, 0.f, 0.f};
    #pragma unroll
    for (int mi = 0; mi < 4; ++mi)
      #pragma unroll
      for (int ni = 0; ni < 4; ++ni) acc[mi][ni] = zero;

    #pragma unroll 1
    for (int ksl = 0; ksl < 8; ++ksl) {       // K slices of 64
      __syncthreads();
      for (int c = tid; c < 128 * 8; c += blockDim.x) {   // stage A slice
        int rr = c >> 3, kc = c & 7;
        *(uint4v*)((char*)ldsA + rr * 144 + kc * 16) =
            *(const uint4v*)((const char*)(zsb + (size_t)(rowbase + rr) * kE +
                                           ksl * 64) + kc * 16);
      }
      // prefetch next A k-slice into cache (global_prefetch_b8 path)
      if (ksl < 7) {
        for (int rr = tid; rr < 128; rr += blockDim.x)
          __builtin_prefetch((const char*)(zsb + (size_t)(rowbase + rr) * kE +
                                           (ksl + 1) * 64), 0, 0);
      }
      __syncthreads();
      #pragma unroll 1
      for (int ks = 0; ks < 2; ++ks) {        // two K=32 WMMA steps per slice
        const int k0 = ks * 32;
        v16bf afr[4];
        #pragma unroll
        for (int mi = 0; mi < 4; ++mi)
          afr[mi] = load_frag(ldsA + (wm * 64 + mi * 16 + mrow) * 72 + k0 + hof);
        #pragma unroll
        for (int ni = 0; ni < 4; ++ni) {
          v16bf b = load_frag(ldsB + (wn * 64 + ni * 16 + mrow) * 520 +
                              ksl * 64 + k0 + hof);
          acc[0][ni] = wmma_bf16(afr[0], b, acc[0][ni]);
          acc[1][ni] = wmma_bf16(afr[1], b, acc[1][ni]);
          acc[2][ni] = wmma_bf16(afr[2], b, acc[2][ni]);
          acc[3][ni] = wmma_bf16(afr[3], b, acc[3][ni]);
        }
      }
    }
    // epilogue: per-row (max, sumexp) over this wave's 64 columns
    #pragma unroll
    for (int mi = 0; mi < 4; ++mi) {
      #pragma unroll
      for (int r = 0; r < 8; ++r) {
        float v0 = acc[mi][0][r] + bv[0];
        float v1 = acc[mi][1][r] + bv[1];
        float v2 = acc[mi][2][r] + bv[2];
        float v3 = acc[mi][3][r] + bv[3];
        float mx = fmaxf(fmaxf(v0, v1), fmaxf(v2, v3));
        mx = half_max16(mx);
        float s = __expf(v0 - mx) + __expf(v1 - mx) +
                  __expf(v2 - mx) + __expf(v3 - mx);
        s = half_sum16(s);
        if ((lane & 15) == 0) {
          int row = rowbase + wm * 64 + mi * 16 + r + hof;
          int pc = chunk * 4 + wn;           // 0..499
          pm[(size_t)pc * kRows + row] = mx;
          ps[(size_t)pc * kRows + row] = s;
        }
      }
    }
  }
}

// ---------------- K5: combine partials + target logit ----------------
__global__ void k_finalize(const float* __restrict__ zs, const int* __restrict__ y,
                           const float* __restrict__ wv, const float* __restrict__ bvoc,
                           const float* __restrict__ pm, const float* __restrict__ ps,
                           float* __restrict__ yp) {
  const int w = threadIdx.x >> 5, lane = threadIdx.x & 31;
  const int row = blockIdx.x * 8 + w;
  if (row >= kRows) return;
  const int yv = y[row];
  const float* a = zs + (size_t)row * kE;
  const float* wr = wv + (size_t)yv * kE;
  float d = 0.f;
  for (int e = lane; e < kE; e += 32) d += a[e] * wr[e];
  d = wave_sum(d);
  float mx = -__builtin_inff();
  for (int c = lane; c < kNPart; c += 32)
    mx = fmaxf(mx, pm[(size_t)c * kRows + row]);
  #pragma unroll
  for (int m = 16; m >= 1; m >>= 1) mx = fmaxf(mx, __shfl_xor(mx, m, 32));
  float s = 0.f;
  for (int c = lane; c < kNPart; c += 32)
    s += ps[(size_t)c * kRows + row] * __expf(pm[(size_t)c * kRows + row] - mx);
  s = wave_sum(s);
  if (lane == 0) yp[row] = d + bvoc[yv] - (mx + __logf(s));
}

// ---------------- host ----------------
static inline size_t alignup(size_t x) { return (x + 255) & ~(size_t)255; }

extern "C" void kernel_launch(void* const* d_in, const int* in_sizes, int n_in,
                              void* d_out, int out_size, void* d_ws, size_t ws_size,
                              hipStream_t stream) {
  const int*   zi     = (const int*)d_in[0];
  const int*   y      = (const int*)d_in[1];
  const float* latent = (const float*)d_in[2];
  const float* W_tr   = (const float*)d_in[3];
  const float* b_tr   = (const float*)d_in[4];
  const float* W_voc  = (const float*)d_in[5];
  const float* b_voc  = (const float*)d_in[6];
  const float* W_ext  = (const float*)d_in[7];
  const float* b_ext  = (const float*)d_in[8];
  float* yp = (float*)d_out;
  char* ws = (char*)d_ws;

  // workspace layout
  size_t off = 0;
  float*  latn   = (float*)(ws + off);  off = alignup(off + (size_t)kB * kS * kE * 4);
  float*  zs_f32 = (float*)(ws + off);  off = alignup(off + (size_t)kRows * kE * 4);
  __bf16* zs_bf  = (__bf16*)(ws + off); off = alignup(off + (size_t)kRows * kE * 2);
  __bf16* wtr_hi = (__bf16*)(ws + off); off = alignup(off + (size_t)kE * kE * 2);
  __bf16* wtr_lo = (__bf16*)(ws + off); off = alignup(off + (size_t)kE * kE * 2);
  __bf16* wv_bf  = (__bf16*)(ws + off); off = alignup(off + (size_t)kV * kE * 2);
  float*  pm     = (float*)(ws + off);  off = alignup(off + (size_t)kNPart * kRows * 4);
  float*  ps     = (float*)(ws + off);  off = alignup(off + (size_t)kNPart * kRows * 4);
  (void)ws_size; (void)in_sizes; (void)n_in; (void)out_size;

  // allow >64KB dynamic LDS (idempotent; ignore errors)
  (void)hipFuncSetAttribute((const void*)k_recurrence,
                            hipFuncAttributeMaxDynamicSharedMemorySize, kRecLds);
  (void)hipFuncSetAttribute((const void*)k_vocab_lse,
                            hipFuncAttributeMaxDynamicSharedMemorySize, kVocLds);

  k_gather_norm<<<kB * kS, 256, 0, stream>>>(zi, latent, latn);
  k_split_bf16<<<(kE * kE + 255) / 256, 256, 0, stream>>>(W_tr, wtr_hi, wtr_lo, kE * kE);
  k_to_bf16<<<2048, 256, 0, stream>>>(W_voc, wv_bf, (size_t)kV * kE);
  k_recurrence<<<2, 512, kRecLds, stream>>>(latn, wtr_hi, wtr_lo, b_tr, W_ext, b_ext,
                                            zs_f32, zs_bf);
  k_vocab_lse<<<2 * kNumChunks, 256, kVocLds, stream>>>(zs_bf, wv_bf, b_voc, pm, ps);
  k_finalize<<<kRows / 8, 256, 0, stream>>>(zs_f32, y, W_voc, b_voc, pm, ps, yp);
}
